// Net_6433861010017
// MI455X (gfx1250) — matrix operand order repaired and checked by reading, not compiled, not run on hardware
//
#include <hip/hip_runtime.h>

#define NT_ 65536   // total nodes (B*N)
#define E_  524288  // total edges
#define B_  32
#define N_  2048
#define F_  128

typedef __attribute__((ext_vector_type(16))) __bf16 v16bf;
typedef __attribute__((ext_vector_type(8)))  float  v8f;

__device__ __forceinline__ unsigned short f2bf(float f) {
  unsigned int u = __float_as_uint(f);
  u += 0x7fffu + ((u >> 16) & 1u);
  return (unsigned short)(u >> 16);
}

// ---------------------------------------------------------------------------
// C[rowBase:rowBase+128, 0:128] = A[rowBase:rowBase+128, 0:128] @ W[128,128]
// bf16 WMMA, f32 accumulate. 256 threads = 8 waves; wave w owns M-tile w.
// LDS holds A and W pre-packed into per-fragment layout so each lane reads
// its 16 bf16 (32B, aligned) with one vector load.
// ---------------------------------------------------------------------------
__global__ __launch_bounds__(256) void gemm_xw(const float* __restrict__ A,
                                               const float* __restrict__ W,
                                               float* __restrict__ C) {
  __shared__ unsigned short lds_a[128 * 128];  // 32 KB, fragment-packed
  __shared__ unsigned short lds_w[128 * 128];  // 32 KB, fragment-packed
  const int tid  = threadIdx.x;
  const int wave = tid >> 5;
  const int lane = tid & 31;
  const long rowBase = (long)blockIdx.x * 128;

  for (int i = tid; i < 128 * 128; i += 256) {
    int r = i >> 7;     // A: row (M) | W: row (K)
    int c = i & 127;    // A: col (K) | W: col (N)
    {  // A fragment pack (ISA 16-bit A 16x32 layout)
      float v = A[(rowBase + r) * F_ + c];
      int kk = c & 31, m = r & 15;
      int ln = m + (((kk >> 3) & 1) << 4);
      int j  = (kk & 7) + ((kk >= 16) ? 8 : 0);
      int fa = ((r >> 4) << 2) + (c >> 5);
      lds_a[fa * 512 + ln * 16 + j] = f2bf(v);
    }
    {  // B fragment pack (same K layout as A: lanes 0-15 K=0..7,16..23, lanes 16-31 K=8..15,24..31)
      float v = W[r * F_ + c];
      int kk = r & 31, nn = c & 15;
      int ln = nn + (((kk >> 3) & 1) << 4);
      int j  = (kk & 7) + ((kk >= 16) ? 8 : 0);
      int fb = ((r >> 5) << 3) + (c >> 4);
      lds_w[fb * 512 + ln * 16 + j] = f2bf(v);
    }
  }
  __syncthreads();

  v8f acc[8] = {};
#pragma unroll
  for (int ks = 0; ks < 4; ++ks) {
    v16bf a = *(const v16bf*)&lds_a[(wave * 4 + ks) * 512 + lane * 16];
#pragma unroll
    for (int n = 0; n < 8; ++n) {
      v16bf b = *(const v16bf*)&lds_w[(ks * 8 + n) * 512 + lane * 16];
      acc[n] = __builtin_amdgcn_wmma_f32_16x16x32_bf16(
          false, a, false, b, (short)0, acc[n], false, false);
    }
  }

  const int colL = lane & 15;
  const int rOff = (lane >> 4) << 3;
#pragma unroll
  for (int n = 0; n < 8; ++n)
#pragma unroll
    for (int r = 0; r < 8; ++r)
      C[(rowBase + wave * 16 + rOff + r) * F_ + n * 16 + colL] = acc[n][r];
}

// ---------------------------------------------------------------------------
// small elementwise / scatter kernels
// ---------------------------------------------------------------------------
__global__ void fill_f32(float* p, float v, long n) {
  long i = (long)blockIdx.x * blockDim.x + threadIdx.x;
  if (i < n) p[i] = v;
}

__global__ void edge_deg(const int* __restrict__ dst, const float* __restrict__ emask,
                         float* __restrict__ deg, int e) {
  int i = blockIdx.x * blockDim.x + threadIdx.x;
  if (i < e) {
    float m = emask[i];
    if (m != 0.f) atomicAdd(&deg[dst[i]], m);
  }
}

__global__ void node_dinv(const float* __restrict__ nmask, float* __restrict__ degdinv, int nt) {
  int i = blockIdx.x * blockDim.x + threadIdx.x;
  if (i < nt) {
    float d = degdinv[i] + nmask[i];
    degdinv[i] = (d > 0.f) ? rsqrtf(d) : 0.f;
  }
}

__global__ void edge_coef(const int* __restrict__ src, const int* __restrict__ dst,
                          const float* __restrict__ emask, const float* __restrict__ dinv,
                          float* __restrict__ coef, int e) {
  int i = blockIdx.x * blockDim.x + threadIdx.x;
  if (i < e) coef[i] = dinv[src[i]] * dinv[dst[i]] * emask[i];
}

// out[n,c] = dinv[n]^2 * nmask[n] * hh[n,c] + b[c]   (self-loop term + bias)
__global__ void self_init(const float* __restrict__ hh, const float* __restrict__ dinv,
                          const float* __restrict__ nmask, const float* __restrict__ b,
                          float* __restrict__ out, long n) {
  long i = (long)blockIdx.x * blockDim.x + threadIdx.x;
  if (i < n) {
    long node = i >> 7;
    int  c    = (int)(i & 127);
    float dv  = dinv[node];
    out[i] = dv * dv * nmask[node] * hh[i] + b[c];
  }
}

// out[dst] += coef[e] * hh[src]   (32 lanes per edge, float4 per lane)
__global__ void edge_agg(const int* __restrict__ src, const int* __restrict__ dst,
                         const float* __restrict__ coef, const float* __restrict__ hh,
                         float* __restrict__ out, int e) {
  long t = (long)blockIdx.x * blockDim.x + threadIdx.x;
  int  ei = (int)(t >> 5);
  int  p  = (int)(t & 31);
  if (ei < e) {
    float cf = coef[ei];
    if (cf != 0.f) {
      int s = src[ei], d = dst[ei];
      const float4 v = *(const float4*)&hh[(long)s * F_ + p * 4];
      float* o = &out[(long)d * F_ + p * 4];
      atomicAdd(&o[0], cf * v.x);
      atomicAdd(&o[1], cf * v.y);
      atomicAdd(&o[2], cf * v.z);
      atomicAdd(&o[3], cf * v.w);
    }
  }
}

__global__ void relu_k(float* p, long n) {
  long i = (long)blockIdx.x * blockDim.x + threadIdx.x;
  if (i < n) p[i] = fmaxf(p[i], 0.f);
}

// sh[n] = h[n,:] . Wp[:,0]
__global__ void gemv_pool(const float* __restrict__ h, const float* __restrict__ Wp,
                          float* __restrict__ sh, int nt) {
  int i = blockIdx.x * blockDim.x + threadIdx.x;
  if (i < nt) {
    const float4* row = (const float4*)&h[(long)i * F_];
    const float4* w   = (const float4*)Wp;
    float s = 0.f;
#pragma unroll 8
    for (int t = 0; t < 32; ++t) {
      float4 a = row[t], b = w[t];
      s += a.x * b.x + a.y * b.y + a.z * b.z + a.w * b.w;
    }
    sh[i] = s;
  }
}

__global__ void edge_agg1(const int* __restrict__ src, const int* __restrict__ dst,
                          const float* __restrict__ coef, const float* __restrict__ sh,
                          float* __restrict__ sagg, int e) {
  int i = blockIdx.x * blockDim.x + threadIdx.x;
  if (i < e) {
    float cf = coef[i];
    if (cf != 0.f) atomicAdd(&sagg[dst[i]], cf * sh[src[i]]);
  }
}

__global__ void score_final(const float* __restrict__ sagg, const float* __restrict__ sh,
                            const float* __restrict__ dinv, const float* __restrict__ nmask,
                            const float* __restrict__ bp, float* __restrict__ score, int nt) {
  int i = blockIdx.x * blockDim.x + threadIdx.x;
  if (i < nt) {
    float dv = dinv[i];
    score[i] = sagg[i] + dv * dv * nmask[i] * sh[i] + bp[0];
  }
}

// Per-graph exact top-k with index-order tie breaking (matches lax.top_k).
__global__ __launch_bounds__(256) void topk_keep(const float* __restrict__ score,
                                                 const float* __restrict__ nmask_in,
                                                 float* __restrict__ keep_out,
                                                 float* __restrict__ gate_out, int k) {
  __shared__ unsigned int keys[N_];
  __shared__ int red[256];
  __shared__ int scanbuf[256];
  const int g = blockIdx.x, tid = threadIdx.x;
  const float* sc = &score[(long)g * N_];
  const float* nm = &nmask_in[(long)g * N_];

  for (int i = tid; i < N_; i += 256) {
    float s = (nm[i] > 0.f) ? sc[i] : -__builtin_inff();
    unsigned u = __float_as_uint(s);
    keys[i] = (u & 0x80000000u) ? ~u : (u | 0x80000000u);  // order-preserving
  }
  __syncthreads();

  // binary search the k-th largest key over the 32-bit key space
  unsigned kth = 0;
  for (int bit = 31; bit >= 0; --bit) {
    unsigned cand = kth | (1u << bit);
    int cnt = 0;
    for (int i = tid; i < N_; i += 256) cnt += (keys[i] >= cand) ? 1 : 0;
    red[tid] = cnt; __syncthreads();
    for (int s2 = 128; s2 > 0; s2 >>= 1) {
      if (tid < s2) red[tid] += red[tid + s2];
      __syncthreads();
    }
    int total = red[0];
    __syncthreads();
    if (total >= k) kth = cand;
  }

  // strictly-greater count
  int cnt = 0;
  for (int i = tid; i < N_; i += 256) cnt += (keys[i] > kth) ? 1 : 0;
  red[tid] = cnt; __syncthreads();
  for (int s2 = 128; s2 > 0; s2 >>= 1) {
    if (tid < s2) red[tid] += red[tid + s2];
    __syncthreads();
  }
  const int gt  = red[0];
  const int rem = k - gt;  // how many ==kth survive (earliest indices)
  __syncthreads();

  // exclusive prefix count of equal keys: each thread owns 8 consecutive idx
  const int base = tid * 8;
  int eq[8]; int local = 0;
#pragma unroll
  for (int j = 0; j < 8; ++j) { eq[j] = local; local += (keys[base + j] == kth) ? 1 : 0; }
  scanbuf[tid] = local; __syncthreads();
  for (int off = 1; off < 256; off <<= 1) {
    int v = (tid >= off) ? scanbuf[tid - off] : 0;
    __syncthreads();
    scanbuf[tid] += v;
    __syncthreads();
  }
  const int excl = scanbuf[tid] - local;

#pragma unroll
  for (int j = 0; j < 8; ++j) {
    int i = base + j;
    unsigned ky = keys[i];
    int kp = (ky > kth) || (ky == kth && (excl + eq[j]) < rem);
    float kf = kp ? 1.f : 0.f;
    keep_out[(long)g * N_ + i] = kf;
    gate_out[(long)g * N_ + i] = kf * tanhf(sc[i]);
  }
}

__global__ void scale_rows(float* __restrict__ h, const float* __restrict__ gate, long n) {
  long i = (long)blockIdx.x * blockDim.x + threadIdx.x;
  if (i < n) h[i] *= gate[i >> 7];
}

__global__ void edge_update(float* __restrict__ emask, const int* __restrict__ src,
                            const int* __restrict__ dst, const float* __restrict__ keep, int e) {
  int i = blockIdx.x * blockDim.x + threadIdx.x;
  if (i < e) emask[i] *= keep[src[i]] * keep[dst[i]];
}

// gvec[g, 0:128]   += max over active nodes
// gvec[g, 128:256] += mean over active nodes (sum / K)
__global__ __launch_bounds__(128) void readout_k(const float* __restrict__ h,
                                                 const float* __restrict__ nmask,
                                                 float invk, float* __restrict__ gvec) {
  int g = blockIdx.x, c = threadIdx.x;
  float mx = -__builtin_inff(), sm = 0.f;
  for (int i = 0; i < N_; ++i) {
    float m = nmask[(long)g * N_ + i];
    float v = h[((long)g * N_ + i) * F_ + c];
    if (m > 0.f) { mx = fmaxf(mx, v); sm += v; }
  }
  gvec[g * 256 + c]       += mx;
  gvec[g * 256 + 128 + c] += sm * invk;
}

// final MLP head + log_softmax (tiny: one workgroup)
__global__ __launch_bounds__(256) void mlp_head(const float* __restrict__ gvec,
                                                const float* __restrict__ W1, const float* __restrict__ b1,
                                                const float* __restrict__ W2, const float* __restrict__ b2,
                                                const float* __restrict__ W3, const float* __restrict__ b3,
                                                float* __restrict__ out) {
  __shared__ float h1[32 * 128];
  __shared__ float h2[32 * 64];
  __shared__ float z[32 * 10];
  const int tid = threadIdx.x;
  for (int o = tid; o < 32 * 128; o += 256) {
    int r = o >> 7, c = o & 127;
    float s = b1[c];
    for (int k = 0; k < 256; ++k) s += gvec[r * 256 + k] * W1[k * 128 + c];
    h1[o] = fmaxf(s, 0.f);
  }
  __syncthreads();
  for (int o = tid; o < 32 * 64; o += 256) {
    int r = o >> 6, c = o & 63;
    float s = b2[c];
    for (int k = 0; k < 128; ++k) s += h1[r * 128 + k] * W2[k * 64 + c];
    h2[o] = fmaxf(s, 0.f);
  }
  __syncthreads();
  for (int o = tid; o < 320; o += 256) {
    int r = o / 10, c = o % 10;
    float s = b3[c];
    for (int k = 0; k < 64; ++k) s += h2[r * 64 + k] * W3[k * 10 + c];
    z[o] = s;
  }
  __syncthreads();
  if (tid < 32) {
    float mx = -__builtin_inff();
    for (int c = 0; c < 10; ++c) mx = fmaxf(mx, z[tid * 10 + c]);
    float se = 0.f;
    for (int c = 0; c < 10; ++c) se += expf(z[tid * 10 + c] - mx);
    float l = logf(se) + mx;
    for (int c = 0; c < 10; ++c) out[tid * 10 + c] = z[tid * 10 + c] - l;
  }
}

// ---------------------------------------------------------------------------
extern "C" void kernel_launch(void* const* d_in, const int* in_sizes, int n_in,
                              void* d_out, int out_size, void* d_ws, size_t ws_size,
                              hipStream_t stream) {
  (void)in_sizes; (void)n_in; (void)out_size; (void)ws_size;

  const float* x   = (const float*)d_in[0];
  const int*   ei  = (const int*)d_in[1];
  const int*   src = ei;
  const int*   dst = ei + E_;
  const float* W1  = (const float*)d_in[3];  const float* b1  = (const float*)d_in[4];
  const float* Wp1 = (const float*)d_in[5];  const float* bp1 = (const float*)d_in[6];
  const float* W2  = (const float*)d_in[7];  const float* b2  = (const float*)d_in[8];
  const float* Wp2 = (const float*)d_in[9];  const float* bp2 = (const float*)d_in[10];
  const float* W3  = (const float*)d_in[11]; const float* b3  = (const float*)d_in[12];
  const float* Wp3 = (const float*)d_in[13]; const float* bp3 = (const float*)d_in[14];
  const float* l1W = (const float*)d_in[15]; const float* l1b = (const float*)d_in[16];
  const float* l2W = (const float*)d_in[17]; const float* l2b = (const float*)d_in[18];
  const float* l3W = (const float*)d_in[19]; const float* l3b = (const float*)d_in[20];

  float* p = (float*)d_ws;
  auto take = [&](size_t nElem) { float* r = p; p += nElem; return r; };
  float* h     = take((size_t)NT_ * F_);   // 32 MB: layer activations (agg output)
  float* hh    = take((size_t)NT_ * F_);   // 32 MB: GEMM output x@W
  float* dinv  = take(NT_);
  float* coef  = take(E_);
  float* emask = take(E_);
  float* nmA   = take(NT_);
  float* nmB   = take(NT_);
  float* sh    = take(NT_);
  float* sagg  = take(NT_);
  float* score = take(NT_);
  float* gate  = take(NT_);
  float* gvec  = take((size_t)B_ * 256);

  const int T = 256;
  auto grid = [&](long n) { return (unsigned)((n + T - 1) / T); };

  fill_f32<<<grid(E_),  T, 0, stream>>>(emask, 1.f, E_);
  fill_f32<<<grid(NT_), T, 0, stream>>>(nmA, 1.f, NT_);
  fill_f32<<<grid(B_ * 256), T, 0, stream>>>(gvec, 0.f, B_ * 256);

  float* nm_cur = nmA;
  float* nm_next = nmB;
  const float* xin = x;

  struct Lyr { const float *W, *b, *Wp, *bp; int K; };
  Lyr ls[3] = { {W1, b1, Wp1, bp1, 1024},
                {W2, b2, Wp2, bp2, 512},
                {W3, b3, Wp3, bp3, 256} };

  for (int l = 0; l < 3; ++l) {
    // GCN conv: hh = xin @ W (WMMA), then normalized aggregation into h
    gemm_xw<<<NT_ / 128, 256, 0, stream>>>(xin, ls[l].W, hh);
    fill_f32<<<grid(NT_), T, 0, stream>>>(dinv, 0.f, NT_);
    edge_deg<<<grid(E_), T, 0, stream>>>(dst, emask, dinv, E_);
    node_dinv<<<grid(NT_), T, 0, stream>>>(nm_cur, dinv, NT_);
    edge_coef<<<grid(E_), T, 0, stream>>>(src, dst, emask, dinv, coef, E_);
    self_init<<<grid((long)NT_ * F_), T, 0, stream>>>(hh, dinv, nm_cur, ls[l].b, h, (long)NT_ * F_);
    edge_agg<<<grid((long)E_ * 32), T, 0, stream>>>(src, dst, coef, hh, h, E_);
    relu_k<<<grid((long)NT_ * F_), T, 0, stream>>>(h, (long)NT_ * F_);

    // SAGPool score = GCNConv(nhid -> 1) with the same dinv/coef
    gemv_pool<<<grid(NT_), T, 0, stream>>>(h, ls[l].Wp, sh, NT_);
    fill_f32<<<grid(NT_), T, 0, stream>>>(sagg, 0.f, NT_);
    edge_agg1<<<grid(E_), T, 0, stream>>>(src, dst, coef, sh, sagg, E_);
    score_final<<<grid(NT_), T, 0, stream>>>(sagg, sh, dinv, nm_cur, ls[l].bp, score, NT_);

    // top-k keep mask + gating, mask updates, readout accumulation
    topk_keep<<<B_, 256, 0, stream>>>(score, nm_cur, nm_next, gate, ls[l].K);
    scale_rows<<<grid((long)NT_ * F_), T, 0, stream>>>(h, gate, (long)NT_ * F_);
    edge_update<<<grid(E_), T, 0, stream>>>(emask, src, dst, nm_next, E_);
    readout_k<<<B_, 128, 0, stream>>>(h, nm_next, 1.f / (float)ls[l].K, gvec);

    float* tmp = nm_cur; nm_cur = nm_next; nm_next = tmp;
    xin = h;
  }

  mlp_head<<<1, 256, 0, stream>>>(gvec, l1W, l1b, l2W, l2b, l3W, l3b, (float*)d_out);
}
